// VQAbstractEncoder_31267361915547
// MI455X (gfx1250) — matrix operand order repaired and checked by reading, not compiled, not run on hardware
//
#include <hip/hip_runtime.h>
#include <hip/hip_bf16.h>

// Problem constants (fixed by the reference).
#define B_ROWS  32768
#define IN_DIM  512
#define D       64
#define NE      8192
#define WROWT   2                        // 16-row tiles per wave
#define MBLK    (8 * 16 * WROWT)         // 256 rows per block
#define NBLK    (B_ROWS / MBLK)          // 128 blocks
#define THREADS 256
#define NT_CODE (NE / 16)                // 512 code tiles

typedef __attribute__((ext_vector_type(16))) __bf16 v16bf;
typedef __attribute__((ext_vector_type(8)))  float  v8f;

union AF {
  v16bf v;
  __bf16 h[16];
  uint4 q[2];
};

// ---------------------------------------------------------------------------
// Prep 1: embed (f32 [NE][D]) -> bf16 copy + f32 squared row norms.
// ---------------------------------------------------------------------------
__global__ void prep_embed(const float* __restrict__ embed,
                           __bf16* __restrict__ e_bf,
                           float* __restrict__ e_norm) {
  int j = blockIdx.x * blockDim.x + threadIdx.x;
  if (j >= NE) return;
  const float4* src = (const float4*)(embed + (size_t)j * D);
  float s = 0.f;
  #pragma unroll
  for (int i = 0; i < D / 4; ++i) {
    float4 v = src[i];
    s += v.x * v.x + v.y * v.y + v.z * v.z + v.w * v.w;
    __bf16* d = e_bf + (size_t)j * D + i * 4;
    d[0] = (__bf16)v.x; d[1] = (__bf16)v.y;
    d[2] = (__bf16)v.z; d[3] = (__bf16)v.w;
  }
  e_norm[j] = s;
}

// ---------------------------------------------------------------------------
// Prep 2: W_body (f32 [K=512][N=64]) -> bf16 in WMMA B-fragment order.
// Fragment tile (kc, nt): lane holds B[k = kc*32 + (lane>>4)*16 + pos][nt*16 + (lane&15)]
// stored at half-index ((kc*4+nt)*32 + lane)*16 + pos  (contiguous k order, same as the A fragments).
// ---------------------------------------------------------------------------
__global__ void prep_w(const float* __restrict__ W, __bf16* __restrict__ Wf) {
  int gid = blockIdx.x * blockDim.x + threadIdx.x;   // 2048 threads
  #pragma unroll
  for (int i = 0; i < 16; ++i) {
    int e = gid + i * 2048;              // 0 .. 32767, coalesced per iteration
    int k = e >> 6, n = e & 63;
    int kc = k >> 5, kin = k & 31;
    int lane = ((kin >> 4) << 4) | (n & 15);
    int pos = kin & 15;
    int nt = n >> 4;
    Wf[((((kc * 4 + nt) * 32) + lane) << 4) + pos] = (__bf16)W[e];
  }
}

// ---------------------------------------------------------------------------
// Main fused kernel: body GEMM -> distance GEMM -> argmin -> gather -> loss.
// ---------------------------------------------------------------------------
__global__ __launch_bounds__(THREADS) void vq_main(
    const float* __restrict__ inputs, const float* __restrict__ bias,
    const float* __restrict__ embed,
    const __bf16* __restrict__ Wf, const __bf16* __restrict__ e_bf,
    const float* __restrict__ e_norm,
    float* __restrict__ out, float* __restrict__ partials) {
  __shared__ __bf16 xs_bf[MBLK * D];   // 32 KB: xs in bf16, row-major per block
  __shared__ int    sidx[MBLK];        // argmin index per row
  __shared__ float  wpart[THREADS / 32];

  const int t = threadIdx.x;
  const int w = t >> 5;                // wave 0..7
  const int l = t & 31;                // lane in wave32
  const int n = l & 15;
  const int half = l >> 4;             // 0 or 1
  const int wbase = w * (16 * WROWT);  // block-local first row of wave
  const int rowbase = blockIdx.x * MBLK + wbase;

  // ---- Phase 1: xs[32x64] = inputs[32x512] @ W[512x64] + b  (bf16 WMMA).
  #pragma unroll
  for (int rt = 0; rt < WROWT; ++rt) {
    v8f cacc[4] = {};
    const float* arow = inputs + (size_t)(rowbase + rt * 16 + n) * IN_DIM;
    #pragma unroll 1
    for (int kc = 0; kc < IN_DIM / 32; ++kc) {
      // A fragment (16-bit 16x32 layout): lane m = l&15,
      // chunk0 = K [kc*32 + half*16 .. +7], chunk1 = chunk0 + 8.
      AF a;
      const int k0 = kc * 32 + half * 16;
      const float4* p0 = (const float4*)(arow + k0);
      const float4* p1 = (const float4*)(arow + k0 + 8);
      float4 x0 = p0[0], x1 = p0[1], y0 = p1[0], y1 = p1[1];
      a.h[0]  = (__bf16)x0.x; a.h[1]  = (__bf16)x0.y;
      a.h[2]  = (__bf16)x0.z; a.h[3]  = (__bf16)x0.w;
      a.h[4]  = (__bf16)x1.x; a.h[5]  = (__bf16)x1.y;
      a.h[6]  = (__bf16)x1.z; a.h[7]  = (__bf16)x1.w;
      a.h[8]  = (__bf16)y0.x; a.h[9]  = (__bf16)y0.y;
      a.h[10] = (__bf16)y0.z; a.h[11] = (__bf16)y0.w;
      a.h[12] = (__bf16)y1.x; a.h[13] = (__bf16)y1.y;
      a.h[14] = (__bf16)y1.z; a.h[15] = (__bf16)y1.w;
      #pragma unroll
      for (int nt = 0; nt < 4; ++nt) {
        AF b;
        const uint4* q = (const uint4*)(Wf + ((((kc * 4 + nt) * 32) + l) << 4));
        b.q[0] = q[0];
        b.q[1] = q[1];
        cacc[nt] = __builtin_amdgcn_wmma_f32_16x16x32_bf16(
            false, a.v, false, b.v, (short)0, cacc[nt], false, false);
      }
    }
    // Bias add + store xs bf16 to LDS (C layout: elem r -> M = r + half*8, N = l&15).
    #pragma unroll
    for (int nt = 0; nt < 4; ++nt) {
      float bb = bias[nt * 16 + n];
      #pragma unroll
      for (int r = 0; r < 8; ++r) {
        int m = r + half * 8;
        xs_bf[(wbase + rt * 16 + m) * D + nt * 16 + n] =
            (__bf16)(cacc[nt][r] + bb);
      }
    }
  }

  // ---- Phase 2: distances to all codes; per-lane running argmin of
  //      d_j = ||e_j||^2 - 2 * xs . e_j   (||xs||^2 is constant per row).
  // Resident A fragments for both row tiles.
  AF a0[WROWT], a1[WROWT];
  #pragma unroll
  for (int rt = 0; rt < WROWT; ++rt) {
    const uint4* src = (const uint4*)(xs_bf + (wbase + rt * 16 + n) * D);
    a0[rt].q[0] = src[2 * half];      // K 0..31:  chunk0 halves = half*16
    a0[rt].q[1] = src[2 * half + 1];  //           chunk1 = +8 halves
    a1[rt].q[0] = src[4 + 2 * half];  // K 32..63
    a1[rt].q[1] = src[4 + 2 * half + 1];
  }

  float best[WROWT * 8];
  int   bidx[WROWT * 8];
  #pragma unroll
  for (int r = 0; r < WROWT * 8; ++r) { best[r] = __builtin_inff(); bidx[r] = 0; }

  // Two code tiles per iteration: each tile's argmin VALU is separated from
  // its producing WMMAs by the other tile's WMMAs (covers the WMMA->VALU
  // co-exec hazard without v_nops), and B fragments are reused across both
  // row tiles.
  #pragma unroll 1
  for (int tt = 0; tt < NT_CODE; tt += 2) {
    AF b0, b1, b2, b3;
    float en0, en1;
    {
      const __bf16* base0 = e_bf + ((size_t)(tt * 16 + n) * D + half * 16);
      const uint4* q0 = (const uint4*)base0;
      const uint4* q1 = (const uint4*)(base0 + 32);
      b0.q[0] = q0[0]; b0.q[1] = q0[1];
      b1.q[0] = q1[0]; b1.q[1] = q1[1];
      const __bf16* base1 = e_bf + ((size_t)((tt + 1) * 16 + n) * D + half * 16);
      const uint4* q2 = (const uint4*)base1;
      const uint4* q3 = (const uint4*)(base1 + 32);
      b2.q[0] = q2[0]; b2.q[1] = q2[1];
      b3.q[0] = q3[0]; b3.q[1] = q3[1];
      en0 = e_norm[tt * 16 + n];
      en1 = e_norm[(tt + 1) * 16 + n];
    }
    if (tt + 16 < NT_CODE)
      __builtin_prefetch(e_bf + (size_t)(tt + 16) * 16 * D, 0, 0);

    v8f c00 = {}, c01 = {}, c10 = {}, c11 = {};
    c00 = __builtin_amdgcn_wmma_f32_16x16x32_bf16(false, a0[0].v, false, b0.v,
                                                  (short)0, c00, false, false);
    c00 = __builtin_amdgcn_wmma_f32_16x16x32_bf16(false, a1[0].v, false, b1.v,
                                                  (short)0, c00, false, false);
    c01 = __builtin_amdgcn_wmma_f32_16x16x32_bf16(false, a0[1].v, false, b0.v,
                                                  (short)0, c01, false, false);
    c01 = __builtin_amdgcn_wmma_f32_16x16x32_bf16(false, a1[1].v, false, b1.v,
                                                  (short)0, c01, false, false);
    c10 = __builtin_amdgcn_wmma_f32_16x16x32_bf16(false, a0[0].v, false, b2.v,
                                                  (short)0, c10, false, false);
    c10 = __builtin_amdgcn_wmma_f32_16x16x32_bf16(false, a1[0].v, false, b3.v,
                                                  (short)0, c10, false, false);
    c11 = __builtin_amdgcn_wmma_f32_16x16x32_bf16(false, a0[1].v, false, b2.v,
                                                  (short)0, c11, false, false);
    c11 = __builtin_amdgcn_wmma_f32_16x16x32_bf16(false, a1[1].v, false, b3.v,
                                                  (short)0, c11, false, false);

    const int j0 = tt * 16 + n;
    const int j1 = (tt + 1) * 16 + n;
    #pragma unroll
    for (int r = 0; r < 8; ++r) {       // tile tt, row-tile 0
      float dval = __builtin_fmaf(-2.f, c00[r], en0);
      bool lt = dval < best[r];
      best[r] = lt ? dval : best[r];
      bidx[r] = lt ? j0 : bidx[r];
    }
    #pragma unroll
    for (int r = 0; r < 8; ++r) {       // tile tt, row-tile 1
      float dval = __builtin_fmaf(-2.f, c01[r], en0);
      bool lt = dval < best[8 + r];
      best[8 + r] = lt ? dval : best[8 + r];
      bidx[8 + r] = lt ? j0 : bidx[8 + r];
    }
    #pragma unroll
    for (int r = 0; r < 8; ++r) {       // tile tt+1, row-tile 0
      float dval = __builtin_fmaf(-2.f, c10[r], en1);
      bool lt = dval < best[r];
      best[r] = lt ? dval : best[r];
      bidx[r] = lt ? j1 : bidx[r];
    }
    #pragma unroll
    for (int r = 0; r < 8; ++r) {       // tile tt+1, row-tile 1
      float dval = __builtin_fmaf(-2.f, c11[r], en1);
      bool lt = dval < best[8 + r];
      best[8 + r] = lt ? dval : best[8 + r];
      bidx[8 + r] = lt ? j1 : bidx[8 + r];
    }
  }

  // ---- Phase 3: reduce argmin across the 16 lanes holding each row.
  #pragma unroll
  for (int rt = 0; rt < WROWT; ++rt) {
    #pragma unroll
    for (int r = 0; r < 8; ++r) {
      float v = best[rt * 8 + r];
      int ix = bidx[rt * 8 + r];
      #pragma unroll
      for (int off = 8; off >= 1; off >>= 1) {
        float ov = __shfl_xor(v, off, 16);
        int oi = __shfl_xor(ix, off, 16);
        if (ov < v || (ov == v && oi < ix)) { v = ov; ix = oi; }
      }
      if (n == 0) sidx[wbase + rt * 16 + r + half * 8] = ix;
    }
  }

  // ---- Phase 4: gather embed rows, write output_x, accumulate loss.
  float lacc = 0.f;
  #pragma unroll 1
  for (int row = 0; row < 16 * WROWT; ++row) {
    int ix = sidx[wbase + row];
    float2 ev = ((const float2*)(embed + (size_t)ix * D))[l];
    int mrow = wbase + row;
    float x0 = (float)xs_bf[mrow * D + l * 2];
    float x1 = (float)xs_bf[mrow * D + l * 2 + 1];
    float d0 = ev.x - x0, d1 = ev.y - x1;
    float2 o;
    o.x = x0 + d0;                     // replicate straight-through arithmetic
    o.y = x1 + d1;
    ((float2*)(out + (size_t)(rowbase + row) * D))[l] = o;
    lacc = __builtin_fmaf(d0, d0, lacc);
    lacc = __builtin_fmaf(d1, d1, lacc);
  }

  // Fixed-order deterministic reduction: wave -> block -> partials[block].
  #pragma unroll
  for (int off = 16; off >= 1; off >>= 1) lacc += __shfl_xor(lacc, off, 32);
  if (l == 0) wpart[w] = lacc;
  __syncthreads();
  if (t == 0) {
    float s = 0.f;
    #pragma unroll
    for (int i = 0; i < THREADS / 32; ++i) s += wpart[i];
    partials[blockIdx.x] = s;
  }
}

// ---------------------------------------------------------------------------
// Final loss reduction (deterministic fixed-order tree over NBLK partials).
// ---------------------------------------------------------------------------
__global__ void loss_reduce(const float* __restrict__ partials,
                            float* __restrict__ out) {
  __shared__ float red[NBLK];
  int t = threadIdx.x;
  red[t] = partials[t];
  __syncthreads();
  for (int s = NBLK / 2; s > 0; s >>= 1) {
    if (t < s) red[t] += red[t + s];
    __syncthreads();
  }
  if (t == 0)
    out[(size_t)B_ROWS * D] = 1.25f * red[0] / (float)((size_t)B_ROWS * D);
}

// ---------------------------------------------------------------------------
extern "C" void kernel_launch(void* const* d_in, const int* in_sizes, int n_in,
                              void* d_out, int out_size, void* d_ws,
                              size_t ws_size, hipStream_t stream) {
  (void)in_sizes; (void)n_in; (void)out_size; (void)ws_size;
  const float* inputs = (const float*)d_in[0];   // [B, IN_DIM] f32
  const float* W      = (const float*)d_in[1];   // [IN_DIM, D] f32
  const float* bias   = (const float*)d_in[2];   // [D] f32
  const float* embed  = (const float*)d_in[3];   // [NE, D] f32
  float* out = (float*)d_out;                    // [B*D] output_x ++ [1] loss

  char* ws = (char*)d_ws;
  __bf16* e_bf    = (__bf16*)ws;                                   // 1 MB
  float*  e_norm  = (float*)(ws + (size_t)NE * D * 2);             // 32 KB
  __bf16* Wf      = (__bf16*)(ws + (size_t)NE * D * 2 + NE * 4);   // 64 KB
  float*  partials =
      (float*)(ws + (size_t)NE * D * 2 + NE * 4 + (size_t)IN_DIM * D * 2);

  prep_embed<<<NE / 256, 256, 0, stream>>>(embed, e_bf, e_norm);
  prep_w<<<8, 256, 0, stream>>>(W, Wf);
  vq_main<<<NBLK, THREADS, 0, stream>>>(inputs, bias, embed, Wf, e_bf, e_norm,
                                        out, partials);
  loss_reduce<<<1, NBLK, 0, stream>>>(partials, out);
}
